// HFMoE_29686813950451
// MI455X (gfx1250) — matrix operand order repaired and checked by reading, not compiled, not run on hardware
//
#include <hip/hip_runtime.h>
#include <hip/hip_bf16.h>

typedef __bf16 bf16;
typedef __attribute__((ext_vector_type(16))) __bf16 v16bf;
typedef __attribute__((ext_vector_type(8)))  __bf16 v8bf;
typedef __attribute__((ext_vector_type(8)))  float  v8f;

#define T_TOK  32768   // B*S = 8*4096
#define H_DIM  512
#define I_DIM  1024
#define E_NUM  8
#define M_TILE 64      // tokens per block in expert kernel
#define MS     (M_TILE / 16)   // 4 M-subtiles
#define NTHR   512
#define NWAVE  (NTHR / 32)     // 16 waves

union AFrag { v16bf v; v8bf h[2]; };

// ---------------------------------------------------------------------------
// 1) Router: per-token logits, top-2 (JAX tie-break: lowest index), softmax
//    over the selected two, append (token,weight) to per-expert lists.
//    Output accumulation later does exactly two nonzero fp32 atomic adds per
//    element onto a zeroed buffer -> commutative -> deterministic.
// ---------------------------------------------------------------------------
__global__ void router_kernel(const float* __restrict__ x,
                              const float* __restrict__ rw,   // [H][E]
                              const float* __restrict__ rb,   // [E]
                              int*   __restrict__ counts,     // [E]
                              int*   __restrict__ ids,        // [E][T]
                              float* __restrict__ wts) {      // [E][T]
  int t = blockIdx.x * blockDim.x + threadIdx.x;
  if (t >= T_TOK) return;
  float lg[E_NUM];
#pragma unroll
  for (int e = 0; e < E_NUM; ++e) lg[e] = rb[e];
  const float* xr = x + (size_t)t * H_DIM;
  for (int k = 0; k < H_DIM; ++k) {
    float xv = xr[k];
    const float* wr = rw + k * E_NUM;
#pragma unroll
    for (int e = 0; e < E_NUM; ++e) lg[e] = fmaf(xv, wr[e], lg[e]);
  }
  int a = 0;
#pragma unroll
  for (int e = 1; e < E_NUM; ++e) if (lg[e] > lg[a]) a = e;
  int b = (a == 0) ? 1 : 0;
#pragma unroll
  for (int e = 0; e < E_NUM; ++e) if (e != a && e != b && lg[e] > lg[b]) b = e;
  // softmax over {lg[a] (max), lg[b]}
  float e2  = __expf(lg[b] - lg[a]);
  float inv = 1.0f / (1.0f + e2);
  float wa = inv, wb = e2 * inv;
  int pa = atomicAdd(&counts[a], 1);
  ids[a * T_TOK + pa] = t; wts[a * T_TOK + pa] = wa;
  int pb = atomicAdd(&counts[b], 1);
  ids[b * T_TOK + pb] = t; wts[b * T_TOK + pb] = wb;
}

// ---------------------------------------------------------------------------
// 2) fp32 -> bf16 copy (activations)
// ---------------------------------------------------------------------------
__global__ void cvt_x_kernel(const float* __restrict__ x, bf16* __restrict__ xb) {
  int i = blockIdx.x * blockDim.x + threadIdx.x;
  xb[i] = (bf16)x[i];
}

// ---------------------------------------------------------------------------
// 3) Batched tiled transpose + fp32->bf16:  src[e][R][C] -> dst[e][C][R]
//    (gives contiguous-K B fragments for WMMA)
// ---------------------------------------------------------------------------
__global__ void transpose_cvt_kernel(const float* __restrict__ src,
                                     bf16* __restrict__ dst, int R, int C) {
  __shared__ float tile[32][33];
  int e = blockIdx.z;
  const float* s = src + (size_t)e * R * C;
  bf16*        d = dst + (size_t)e * R * C;
  int c0 = blockIdx.x * 32, r0 = blockIdx.y * 32;
  int tx = threadIdx.x, ty = threadIdx.y;        // 32 x 8
#pragma unroll
  for (int i = 0; i < 32; i += 8)
    tile[ty + i][tx] = s[(size_t)(r0 + ty + i) * C + (c0 + tx)];
  __syncthreads();
#pragma unroll
  for (int i = 0; i < 32; i += 8)
    d[(size_t)(c0 + ty + i) * R + (r0 + tx)] = (bf16)tile[tx][ty + i];
}

// ---------------------------------------------------------------------------
// 4) Fused expert kernel: gather 64 tokens -> GEMM1(bf16 WMMA) -> quickGELU
//    -> act in LDS -> GEMM2(bf16 WMMA) -> bias, row-scale, atomic scatter-add.
//    M_TILE=64 halves per-FLOP L2 weight traffic vs 32 (each B fragment feeds
//    4 WMMAs). LDS: 64KB x-tile + 128KB act-tile = 192KB of 320KB per WGP.
// ---------------------------------------------------------------------------
__launch_bounds__(NTHR)
__global__ void expert_kernel(const bf16*  __restrict__ xb,     // [T][H] bf16
                              const bf16*  __restrict__ wguT,   // [E][2I][H] bf16
                              const float* __restrict__ bgu,    // [E][2I]
                              const bf16*  __restrict__ wdT,    // [E][H][I] bf16
                              const float* __restrict__ bd,     // [E][H]
                              const int*   __restrict__ ids,
                              const float* __restrict__ wts,
                              const int*   __restrict__ counts,
                              float*       __restrict__ out) {  // [T][H] fp32
  __shared__ __align__(16) bf16 x_lds[M_TILE][H_DIM];    // 64 KB
  __shared__ __align__(16) bf16 act_lds[M_TILE][I_DIM];  // 128 KB
  __shared__ int   tok_s[M_TILE];
  __shared__ float wt_s[M_TILE];

  const int e    = blockIdx.y;
  const int cnt  = counts[e];
  const int base = blockIdx.x * M_TILE;
  if (base >= cnt) return;

  const int tid = threadIdx.x;
  if (tid < M_TILE) {
    int idx = base + tid;
    int src = (idx < cnt) ? idx : base;                 // pad with a valid row
    tok_s[tid] = ids[e * T_TOK + src];
    wt_s[tid]  = (idx < cnt) ? wts[e * T_TOK + idx] : 0.0f;  // zero kills pads
  }
  __syncthreads();

  // Gather x rows (bf16) into LDS, 16B chunks
  {
    const int chunks = M_TILE * H_DIM / 8;
    for (int c = tid; c < chunks; c += NTHR) {
      int row = c / (H_DIM / 8);
      int col = (c % (H_DIM / 8)) * 8;
      *(v8bf*)&x_lds[row][col] =
          *(const v8bf*)&xb[(size_t)tok_s[row] * H_DIM + col];
    }
  }
  __syncthreads();

  const int wave = tid >> 5, lane = tid & 31;
  const int hi = lane >> 4, ln = lane & 15;
  const v8f vzero = {0.f, 0.f, 0.f, 0.f, 0.f, 0.f, 0.f, 0.f};

  const bf16*  wgu_e = wguT + (size_t)e * 2 * I_DIM * H_DIM;
  const float* bgu_e = bgu  + (size_t)e * 2 * I_DIM;

  // ---- GEMM1 + quickGELU: act[64][I] = f(x @ Wgu + bgu) ----
  for (int nt = wave; nt < I_DIM / 16; nt += NWAVE) {
    const int nb = nt * 16;
    v8f cg[MS], cu[MS];
#pragma unroll
    for (int m = 0; m < MS; ++m) { cg[m] = vzero; cu[m] = vzero; }
    for (int kb = 0; kb < H_DIM; kb += 32) {
      // B fragments: 32B contiguous per lane from transposed weights
      // (lanes 0-15: K=kb..kb+15, lanes 16-31: K=kb+16..kb+31, N=lane%16)
      v16bf bg = *(const v16bf*)&wgu_e[(size_t)(nb + ln) * H_DIM + kb + 16 * hi];
      v16bf bu = *(const v16bf*)&wgu_e[(size_t)(I_DIM + nb + ln) * H_DIM + kb + 16 * hi];
#pragma unroll
      for (int m = 0; m < MS; ++m) {
        // A fragment (16-bit 16x32): low lanes K[0..15],
        // high lanes K[16..31], M = lane%16
        AFrag a;
        a.h[0] = *(const v8bf*)&x_lds[m * 16 + ln][kb + 16 * hi];
        a.h[1] = *(const v8bf*)&x_lds[m * 16 + ln][kb + 16 * hi + 8];
        cg[m] = __builtin_amdgcn_wmma_f32_16x16x32_bf16(false, a.v, false, bg,
                                                        (short)0, cg[m], false, false);
        cu[m] = __builtin_amdgcn_wmma_f32_16x16x32_bf16(false, a.v, false, bu,
                                                        (short)0, cu[m], false, false);
      }
    }
    const float bgate = bgu_e[nb + ln];
    const float bup   = bgu_e[I_DIM + nb + ln];
#pragma unroll
    for (int m = 0; m < MS; ++m)
#pragma unroll
      for (int r = 0; r < 8; ++r) {
        float g = cg[m][r] + bgate;
        float u = cu[m][r] + bup;
        float s = 1.0f / (1.0f + __expf(-1.702f * g));     // sigmoid
        float act = (u + 1.0f) * (g * s);                  // (up+1)*quickGELU(gate)
        act_lds[m * 16 + r + 8 * hi][nb + ln] = (bf16)act; // C: M=r+8*(lane>=16)
      }
  }
  __syncthreads();

  const bf16*  wd_e = wdT + (size_t)e * H_DIM * I_DIM;
  const float* bd_e = bd  + (size_t)e * H_DIM;

  // ---- GEMM2: out += w_tok * (act @ Wd + bd) ----
  for (int nt = wave; nt < H_DIM / 16; nt += NWAVE) {
    const int nb = nt * 16;
    v8f c[MS];
#pragma unroll
    for (int m = 0; m < MS; ++m) c[m] = vzero;
    for (int kb = 0; kb < I_DIM; kb += 32) {
      v16bf b = *(const v16bf*)&wd_e[(size_t)(nb + ln) * I_DIM + kb + 16 * hi];
#pragma unroll
      for (int m = 0; m < MS; ++m) {
        AFrag a;
        a.h[0] = *(const v8bf*)&act_lds[m * 16 + ln][kb + 16 * hi];
        a.h[1] = *(const v8bf*)&act_lds[m * 16 + ln][kb + 16 * hi + 8];
        c[m] = __builtin_amdgcn_wmma_f32_16x16x32_bf16(false, a.v, false, b,
                                                       (short)0, c[m], false, false);
      }
    }
    const float bias = bd_e[nb + ln];
#pragma unroll
    for (int m = 0; m < MS; ++m)
#pragma unroll
      for (int r = 0; r < 8; ++r) {
        int row = m * 16 + r + 8 * hi;
        float val = (c[m][r] + bias) * wt_s[row];
        atomicAdd(&out[(size_t)tok_s[row] * H_DIM + (nb + ln)], val);
      }
  }
}

// ---------------------------------------------------------------------------
extern "C" void kernel_launch(void* const* d_in, const int* in_sizes, int n_in,
                              void* d_out, int out_size, void* d_ws, size_t ws_size,
                              hipStream_t stream) {
  const float* x   = (const float*)d_in[0];   // [B,S,H]
  const float* rw  = (const float*)d_in[1];   // [H,E]
  const float* rb  = (const float*)d_in[2];   // [E]
  const float* wgu = (const float*)d_in[3];   // [E,H,2I]
  const float* bgu = (const float*)d_in[4];   // [E,2I]
  const float* wd  = (const float*)d_in[5];   // [E,I,H]
  const float* bd  = (const float*)d_in[6];   // [E,H]
  float* out = (float*)d_out;

  char* ws = (char*)d_ws;
  bf16* xb    = (bf16*)ws;  ws += (size_t)T_TOK * H_DIM * sizeof(bf16);
  bf16* wguT  = (bf16*)ws;  ws += (size_t)E_NUM * 2 * I_DIM * H_DIM * sizeof(bf16);
  bf16* wdT   = (bf16*)ws;  ws += (size_t)E_NUM * H_DIM * I_DIM * sizeof(bf16);
  int*  ids   = (int*)ws;   ws += (size_t)E_NUM * T_TOK * sizeof(int);
  float* wts  = (float*)ws; ws += (size_t)E_NUM * T_TOK * sizeof(float);
  int*  counts= (int*)ws;

  hipMemsetAsync(counts, 0, E_NUM * sizeof(int), stream);
  hipMemsetAsync(out, 0, (size_t)T_TOK * H_DIM * sizeof(float), stream);

  cvt_x_kernel<<<(T_TOK * H_DIM) / 256, 256, 0, stream>>>(x, xb);
  transpose_cvt_kernel<<<dim3(2 * I_DIM / 32, H_DIM / 32, E_NUM), dim3(32, 8), 0, stream>>>(
      wgu, wguT, H_DIM, 2 * I_DIM);
  transpose_cvt_kernel<<<dim3(H_DIM / 32, I_DIM / 32, E_NUM), dim3(32, 8), 0, stream>>>(
      wd, wdT, I_DIM, H_DIM);
  router_kernel<<<T_TOK / 256, 256, 0, stream>>>(x, rw, rb, counts, ids, wts);
  expert_kernel<<<dim3(T_TOK / M_TILE, E_NUM), NTHR, 0, stream>>>(
      xb, wguT, bgu, wdT, bd, ids, wts, counts, out);
}
